// BiLSTM_CRF_47296179863843
// MI455X (gfx1250) — matrix operand order repaired and checked by reading, not compiled, not run on hardware
//
#include <hip/hip_runtime.h>

// ---------------------------------------------------------------------------
// BiLSTM-CRF for MI455X (gfx1250, wave32, WMMA bf16)
//   prep_kernel : f32 -> bf16 weight/embedding conversion, bias fuse, zeroing
//   lstm_kernel : 8 blocks (2 dirs x 4 hidden-slices). Weights resident in
//                 320KB LDS, cell state in VGPRs, h exchanged via global
//                 ping-pong staged with GLOBAL_LOAD_ASYNC_TO_LDS_B128.
//                 Emits v_wmma_f32_16x16x32_bf16.
//   crf_kernel  : Viterbi forward scan + backtrace (1 block, 128 lanes).
// ---------------------------------------------------------------------------

#define TT    512
#define BB    128
#define EE    256
#define HH    256
#define VV    50000
#define NTAGC 12
#define STARTT 10
#define STOPT  11
#define NEGV  (-10000.0f)

typedef __attribute__((ext_vector_type(16))) __bf16 v16bf;
typedef __attribute__((ext_vector_type(8)))  float  v8f;

union FragU { uint4 q[2]; v16bf v; };

__device__ __forceinline__ unsigned short f2bf(float f) {
    unsigned u = __builtin_bit_cast(unsigned, f);
    unsigned r = u + 0x7FFFu + ((u >> 16) & 1u);          // round-to-nearest-even
    return (unsigned short)(r >> 16);
}
__device__ __forceinline__ float bf2f(unsigned short h) {
    unsigned u = ((unsigned)h) << 16;
    return __builtin_bit_cast(float, u);
}
__device__ __forceinline__ float sigm(float x)  { return 1.0f / (1.0f + __expf(-x)); }
__device__ __forceinline__ float tanhf_(float x){ return 2.0f / (1.0f + __expf(-2.0f * x)) - 1.0f; }

__device__ __forceinline__ v8f wmma_bf(v16bf a, v16bf b, v8f c) {
    return __builtin_amdgcn_wmma_f32_16x16x32_bf16(false, a, false, b, (short)0, c, false, false);
}

// A-fragment (16x32, M rows = row0..row0+15): lane m=row0+(lane&15);
// K = kb + {0..7,16..23} (lanes 0-15) / {8..15,24..31} (lanes 16-31)
__device__ __forceinline__ v16bf fragA_lds(const unsigned short* base, int row0, int kb, int lane) {
    const unsigned short* p = base + (row0 + (lane & 15)) * 256 + kb + ((lane >> 4) << 3);
    FragU f;
    f.q[0] = *reinterpret_cast<const uint4*>(p);
    f.q[1] = *reinterpret_cast<const uint4*>(p + 16);
    return f.v;
}
__device__ __forceinline__ v16bf fragA_row(const unsigned short* rowptr, int kb, int lane) {
    const unsigned short* p = rowptr + kb + ((lane >> 4) << 3);
    FragU f;
    f.q[0] = *reinterpret_cast<const uint4*>(p);
    f.q[1] = *reinterpret_cast<const uint4*>(p + 16);
    return f.v;
}
// B-fragment (32x16, columns = W rows row0..row0+15, column-major):
// lane n=row0+(lane&15); K = kb + {0..7,16..23} (lanes 0-15) / {8..15,24..31} (lanes 16-31)
__device__ __forceinline__ v16bf fragB_any(const unsigned short* base, int row0, int kb, int lane) {
    const unsigned short* p = base + (row0 + (lane & 15)) * 256 + kb + ((lane >> 4) << 3);
    FragU f;
    f.q[0] = *reinterpret_cast<const uint4*>(p);
    f.q[1] = *reinterpret_cast<const uint4*>(p + 16);
    return f.v;
}

// Async-copy 64KB (h state) global -> LDS via the CDNA5 async-to-LDS path.
// 512 threads x 8 chunks x 16B. Each wave waits its own ASYNCcnt; the
// following __syncthreads() publishes all slices to the workgroup.
__device__ __forceinline__ void async_copy_h(const unsigned short* hcur,
                                             unsigned hstage_lds_off, int tid) {
    #pragma unroll
    for (int i = 0; i < 8; ++i) {
        int idx = tid + i * 512;                       // 16B chunk index (0..4095)
        unsigned ldsa = hstage_lds_off + idx * 16;     // LDS byte address (wave-relative)
        const void* g = (const char*)hcur + (size_t)idx * 16;
        asm volatile("global_load_async_to_lds_b128 %0, %1, off"
                     :: "v"(ldsa), "v"(g) : "memory");
    }
    asm volatile("s_wait_asynccnt 0x0" ::: "memory");
}

// feats tile: h(128x256) @ Wout^T(256x16), one 16-row M-tile
__device__ __forceinline__ void feats_gemm(const unsigned short* hstage,
                                           const unsigned short* Wout,
                                           float* feats, int tp, int mtile, int lane) {
    v8f acc = {0.f, 0.f, 0.f, 0.f, 0.f, 0.f, 0.f, 0.f};
    #pragma unroll
    for (int kb = 0; kb < 256; kb += 32) {
        v16bf a = fragA_lds(hstage, mtile * 16, kb, lane);
        v16bf b = fragB_any(Wout, 0, kb, lane);
        acc = wmma_bf(a, b, acc);
    }
    int col = lane & 15;
    int mb  = mtile * 16 + ((lane >> 4) << 3);
    #pragma unroll
    for (int r = 0; r < 8; ++r)
        feats[(tp * BB + mb + r) * 16 + col] = acc[r];
}

// ---------------------------------------------------------------------------
__global__ void prep_kernel(const float* __restrict__ emb,
                            const float* __restrict__ Wih_f, const float* __restrict__ Whh_f,
                            const float* __restrict__ bih_f, const float* __restrict__ bhh_f,
                            const float* __restrict__ Wih_b, const float* __restrict__ Whh_b,
                            const float* __restrict__ bih_b, const float* __restrict__ bhh_b,
                            const float* __restrict__ W_out,
                            unsigned short* embbf,
                            unsigned short* WihF, unsigned short* WhhF,
                            unsigned short* WihB, unsigned short* WhhB,
                            unsigned short* WoutF, unsigned short* WoutB,
                            float* biasF, float* biasB,
                            unsigned short* hbuf, unsigned int* counters) {
    long i = (long)blockIdx.x * blockDim.x + threadIdx.x;
    long stride = (long)gridDim.x * blockDim.x;
    for (long k = i; k < (long)VV * EE; k += stride) embbf[k] = f2bf(emb[k]);
    for (long k = i; k < 262144; k += stride) {
        WihF[k] = f2bf(Wih_f[k]); WhhF[k] = f2bf(Whh_f[k]);
        WihB[k] = f2bf(Wih_b[k]); WhhB[k] = f2bf(Whh_b[k]);
        if (k < 4096) {
            int r = (int)(k >> 8), c = (int)(k & 255);
            WoutF[k] = (r < NTAGC) ? f2bf(W_out[r * 512 + c])       : (unsigned short)0;
            WoutB[k] = (r < NTAGC) ? f2bf(W_out[r * 512 + 256 + c]) : (unsigned short)0;
        }
        if (k < 1024) { biasF[k] = bih_f[k] + bhh_f[k]; biasB[k] = bih_b[k] + bhh_b[k]; }
    }
    for (long k = i; k < 4 * BB * HH; k += stride) hbuf[k] = 0;
    if (i < 8) counters[i] = 0u;
}

// ---------------------------------------------------------------------------
// grid = 8 blocks: blockIdx>>2 = direction, blockIdx&3 = hidden-unit slice d.
// LDS: Wih slice [256][256] bf16 (128KB) + Whh slice (128KB) + h stage (64KB) = 320KB
__global__ __launch_bounds__(512, 1)
void lstm_kernel(const int* __restrict__ sent, const float* __restrict__ maskG,
                 const unsigned short* __restrict__ embbf,
                 const unsigned short* __restrict__ WihF, const unsigned short* __restrict__ WhhF,
                 const unsigned short* __restrict__ WihB, const unsigned short* __restrict__ WhhB,
                 const unsigned short* __restrict__ WoutF, const unsigned short* __restrict__ WoutB,
                 const float* __restrict__ biasF, const float* __restrict__ biasB,
                 unsigned short* __restrict__ hbuf,
                 float* __restrict__ featsF, float* __restrict__ featsB,
                 unsigned int* __restrict__ counters) {
    extern __shared__ char smem[];
    unsigned short* WihL   = (unsigned short*)smem;       // 256x256 bf16
    unsigned short* WhhL   = WihL + 256 * 256;            // 256x256 bf16
    unsigned short* hstage = WhhL + 256 * 256;            // 128x256 bf16
    const unsigned hstage_off = (unsigned)(unsigned long long)hstage;  // LDS byte offset

    const int dir = blockIdx.x >> 2;
    const int d   = blockIdx.x & 3;
    const unsigned short* Wih  = dir ? WihB  : WihF;
    const unsigned short* Whh  = dir ? WhhB  : WhhF;
    const unsigned short* Wout = dir ? WoutB : WoutF;
    const float* bias          = dir ? biasB : biasF;
    float* feats               = dir ? featsB : featsF;
    unsigned short* hb0 = hbuf + dir * 2 * (BB * HH);
    unsigned short* hb1 = hb0 + BB * HH;
    unsigned int* cnt = counters + dir;

    const int tid  = threadIdx.x;
    const int lane = tid & 31;
    const int wv   = tid >> 5;            // 16 waves
    const int ln   = lane & 15;
    const int hi   = lane >> 4;

    // --- stage weight slices (rows g*256 + d*64 + jj -> local row g*64+jj) ---
    for (int i = tid; i < 8192; i += 512) {           // 256 rows * 32 uint4 per matrix
        int lr = i >> 5, q = i & 31;
        int g = lr >> 6, jj = lr & 63;
        int n = g * 256 + d * 64 + jj;
        ((uint4*)WihL)[i] = ((const uint4*)(Wih + n * 256))[q];
        ((uint4*)WhhL)[i] = ((const uint4*)(Whh + n * 256))[q];
    }

    // wave tile assignment: j-tile jt (16 local cols), two m-tiles mt0/mt1
    const int jt    = wv & 3;
    const int mt0   = wv >> 2;            // 0..3
    const int mt1   = mt0 + 4;            // 4..7
    const int jglob = d * 64 + jt * 16 + ln;   // hidden col 0..255

    float bias_g[4];
    #pragma unroll
    for (int g = 0; g < 4; ++g) bias_g[g] = bias[g * 256 + jglob];

    float cR[2][8];
    #pragma unroll
    for (int it = 0; it < 2; ++it)
        #pragma unroll
        for (int r = 0; r < 8; ++r) cR[it][r] = 0.f;

    __syncthreads();

    for (int s = 0; s < TT; ++s) {
        const int tt = dir ? (TT - 1 - s) : s;
        const unsigned short* hcur = (s & 1) ? hb1 : hb0;
        unsigned short*       hnxt = (s & 1) ? hb0 : hb1;

        // stage full h into LDS: direct L2->LDS async copy (ASYNCcnt path)
        async_copy_h(hcur, hstage_off, tid);
        __syncthreads();

        // output projection for the previous step's h (M-partitioned, no atomics)
        if (s > 0 && wv < 2) {
            int tp = dir ? (TT - s) : (s - 1);
            feats_gemm(hstage, Wout, feats, tp, 2 * d + wv, lane);
        }

        // --- gate GEMMs: g = x_t @ Wih^T + h @ Whh^T + bias ---
        v8f acc[2][4];
        #pragma unroll
        for (int it = 0; it < 2; ++it)
            #pragma unroll
            for (int g = 0; g < 4; ++g)
                #pragma unroll
                for (int r = 0; r < 8; ++r) acc[it][g][r] = bias_g[g];

        const int tok0 = sent[tt * BB + mt0 * 16 + ln];
        const int tok1 = sent[tt * BB + mt1 * 16 + ln];
        const unsigned short* xr0 = embbf + tok0 * EE;
        const unsigned short* xr1 = embbf + tok1 * EE;

        // prefetch next step's embedding rows (global_prefetch_b8)
        if (s + 1 < TT) {
            const int tn = dir ? (TT - 2 - s) : (s + 1);
            __builtin_prefetch(embbf + (size_t)sent[tn * BB + mt0 * 16 + ln] * EE, 0, 1);
            __builtin_prefetch(embbf + (size_t)sent[tn * BB + mt1 * 16 + ln] * EE, 0, 1);
        }

        #pragma unroll
        for (int kb = 0; kb < 256; kb += 32) {
            v16bf ax0 = fragA_row(xr0, kb, lane);
            v16bf ax1 = fragA_row(xr1, kb, lane);
            v16bf ah0 = fragA_lds(hstage, mt0 * 16, kb, lane);
            v16bf ah1 = fragA_lds(hstage, mt1 * 16, kb, lane);
            #pragma unroll
            for (int g = 0; g < 4; ++g) {
                v16bf bi = fragB_any(WihL, g * 64 + jt * 16, kb, lane);
                v16bf bh = fragB_any(WhhL, g * 64 + jt * 16, kb, lane);
                acc[0][g] = wmma_bf(ax0, bi, acc[0][g]);
                acc[0][g] = wmma_bf(ah0, bh, acc[0][g]);
                acc[1][g] = wmma_bf(ax1, bi, acc[1][g]);
                acc[1][g] = wmma_bf(ah1, bh, acc[1][g]);
            }
        }

        // --- LSTM cell update; c lives in VGPRs for the whole sequence ---
        #pragma unroll
        for (int it = 0; it < 2; ++it) {
            const int mbase = (it ? mt1 : mt0) * 16 + hi * 8;
            #pragma unroll
            for (int r = 0; r < 8; ++r) {
                int m = mbase + r;
                float iv = sigm(acc[it][0][r]);
                float fv = sigm(acc[it][1][r]);
                float gv = tanhf_(acc[it][2][r]);
                float ov = sigm(acc[it][3][r]);
                float cold = cR[it][r];
                float c2 = fv * cold + iv * gv;
                float h2 = ov * tanhf_(c2);
                float mk = maskG[tt * BB + m];
                float hold = bf2f(hstage[m * 256 + jglob]);
                h2 = mk * h2 + (1.f - mk) * hold;
                c2 = mk * c2 + (1.f - mk) * cold;
                cR[it][r] = c2;
                hnxt[m * 256 + jglob] = f2bf(h2);
            }
        }

        // --- 4-block (per-direction) step barrier: h2 must be globally visible ---
        __syncthreads();
        if (tid == 0) {
            __threadfence();
            atomicAdd(cnt, 1u);
            unsigned target = 4u * (unsigned)(s + 1);
            while (__hip_atomic_load(cnt, __ATOMIC_ACQUIRE, __HIP_MEMORY_SCOPE_AGENT) < target)
                __builtin_amdgcn_s_sleep(1);
        }
        __syncthreads();
    }

    // epilogue: projection for the final step's h
    {
        const unsigned short* hcur = (TT & 1) ? hb1 : hb0;
        async_copy_h(hcur, hstage_off, tid);
        __syncthreads();
        if (wv < 2) {
            int tp = dir ? 0 : (TT - 1);
            feats_gemm(hstage, Wout, feats, tp, 2 * d + wv, lane);
        }
    }
}

// ---------------------------------------------------------------------------
__global__ __launch_bounds__(128, 1)
void crf_kernel(const float* __restrict__ featsF, const float* __restrict__ featsB,
                const float* __restrict__ b_out, const float* __restrict__ trans,
                const float* __restrict__ maskG,
                unsigned char* __restrict__ ptrs, float* __restrict__ out) {
    __shared__ float sT[NTAGC * NTAGC];
    __shared__ float sB[NTAGC];
    int tid = threadIdx.x;
    for (int i = tid; i < NTAGC * NTAGC; i += 128) sT[i] = trans[i];
    if (tid < NTAGC) sB[tid] = b_out[tid];
    __syncthreads();

    const int b = tid;     // 128 threads == B
    float sc[NTAGC];
    #pragma unroll
    for (int j = 0; j < NTAGC; ++j) sc[j] = NEGV;
    sc[STARTT] = 0.f;

    for (int t = 0; t < TT; ++t) {
        float mk = maskG[t * BB + b];
        const float* fF = featsF + (t * BB + b) * 16;
        const float* fB = featsB + (t * BB + b) * 16;
        float ns[NTAGC];
        #pragma unroll
        for (int i = 0; i < NTAGC; ++i) {
            float best = sc[0] + sT[i * NTAGC + 0];
            int bj = 0;
            #pragma unroll
            for (int j = 1; j < NTAGC; ++j) {
                float v = sc[j] + sT[i * NTAGC + j];
                if (v > best) { best = v; bj = j; }
            }
            float feat = (fF[i] + fB[i] + sB[i]) * mk;
            ns[i] = best + feat;
            ptrs[(t * BB + b) * 16 + i] = (unsigned char)bj;
        }
        #pragma unroll
        for (int i = 0; i < NTAGC; ++i) sc[i] = mk * ns[i] + (1.f - mk) * sc[i];
    }

    #pragma unroll
    for (int j = 0; j < NTAGC; ++j) sc[j] += sT[STOPT * NTAGC + j];
    float best = sc[0]; int bt = 0;
    #pragma unroll
    for (int j = 1; j < NTAGC; ++j) if (sc[j] > best) { best = sc[j]; bt = j; }
    out[TT * BB + b] = best;

    int tag = bt;
    for (int t = TT - 1; t >= 0; --t) {
        float mk = maskG[t * BB + b];
        bool valid = mk > 0.5f;
        out[t * BB + b] = valid ? (float)tag : -1.0f;
        int prev = (int)ptrs[(t * BB + b) * 16 + tag];
        tag = valid ? prev : tag;
    }
}

// ---------------------------------------------------------------------------
extern "C" void kernel_launch(void* const* d_in, const int* in_sizes, int n_in,
                              void* d_out, int out_size, void* d_ws, size_t ws_size,
                              hipStream_t stream) {
    const int*   sentence = (const int*)  d_in[0];
    const float* mask     = (const float*)d_in[1];
    const float* emb      = (const float*)d_in[2];
    const float* Wih_f    = (const float*)d_in[3];
    const float* Whh_f    = (const float*)d_in[4];
    const float* bih_f    = (const float*)d_in[5];
    const float* bhh_f    = (const float*)d_in[6];
    const float* Wih_b    = (const float*)d_in[7];
    const float* Whh_b    = (const float*)d_in[8];
    const float* bih_b    = (const float*)d_in[9];
    const float* bhh_b    = (const float*)d_in[10];
    const float* W_out    = (const float*)d_in[11];
    const float* b_out    = (const float*)d_in[12];
    const float* trans    = (const float*)d_in[13];

    char* ws = (char*)d_ws;
    const size_t OFF_EMB   = 0;                          // 50000*256*2
    const size_t OFF_WIHF  = OFF_EMB   + (size_t)VV * EE * 2;
    const size_t OFF_WHHF  = OFF_WIHF  + 524288;
    const size_t OFF_WIHB  = OFF_WHHF  + 524288;
    const size_t OFF_WHHB  = OFF_WIHB  + 524288;
    const size_t OFF_WOUTF = OFF_WHHB  + 524288;
    const size_t OFF_WOUTB = OFF_WOUTF + 8192;
    const size_t OFF_BIASF = OFF_WOUTB + 8192;
    const size_t OFF_BIASB = OFF_BIASF + 4096;
    const size_t OFF_HBUF  = OFF_BIASB + 4096;           // 4*32768*2
    const size_t OFF_FEATF = OFF_HBUF  + 524288;         // T*B*16*4
    const size_t OFF_FEATB = OFF_FEATF + 4194304;
    const size_t OFF_PTRS  = OFF_FEATB + 4194304;        // T*B*16
    const size_t OFF_CNT   = OFF_PTRS  + 1048576;

    unsigned short* embbf = (unsigned short*)(ws + OFF_EMB);
    unsigned short* WihF  = (unsigned short*)(ws + OFF_WIHF);
    unsigned short* WhhF  = (unsigned short*)(ws + OFF_WHHF);
    unsigned short* WihB  = (unsigned short*)(ws + OFF_WIHB);
    unsigned short* WhhB  = (unsigned short*)(ws + OFF_WHHB);
    unsigned short* WoutF = (unsigned short*)(ws + OFF_WOUTF);
    unsigned short* WoutB = (unsigned short*)(ws + OFF_WOUTB);
    float*          biasF = (float*)(ws + OFF_BIASF);
    float*          biasB = (float*)(ws + OFF_BIASB);
    unsigned short* hbuf  = (unsigned short*)(ws + OFF_HBUF);
    float*          featF = (float*)(ws + OFF_FEATF);
    float*          featB = (float*)(ws + OFF_FEATB);
    unsigned char*  ptrs  = (unsigned char*)(ws + OFF_PTRS);
    unsigned int*   cnt   = (unsigned int*)(ws + OFF_CNT);

    prep_kernel<<<4096, 256, 0, stream>>>(emb, Wih_f, Whh_f, bih_f, bhh_f,
                                          Wih_b, Whh_b, bih_b, bhh_b, W_out,
                                          embbf, WihF, WhhF, WihB, WhhB,
                                          WoutF, WoutB, biasF, biasB, hbuf, cnt);

    lstm_kernel<<<8, 512, 327680, stream>>>(sentence, mask, embbf,
                                            WihF, WhhF, WihB, WhhB,
                                            WoutF, WoutB, biasF, biasB,
                                            hbuf, featF, featB, cnt);

    crf_kernel<<<1, 128, 0, stream>>>(featF, featB, b_out, trans, mask,
                                      ptrs, (float*)d_out);
}
